// JointCA_15298673508694
// MI455X (gfx1250) — compile-verified, matrix operand order checked by reading, not run on hardware
//
#include <hip/hip_runtime.h>
#include <hip/hip_bf16.h>
#include <stdint.h>

typedef __attribute__((ext_vector_type(16))) __bf16 v16bf;
typedef __attribute__((ext_vector_type(8)))  __bf16 v8bf;
typedef __attribute__((ext_vector_type(8)))  float  v8f;
typedef __attribute__((ext_vector_type(4)))  float  vf4;

#define NB 64
#define NN 4096
#define CC 768
#define HH 12
#define TT 10
#define DD 64
#define BHC (NB*HH)

// ---- workspace layout (bytes) ----
#define WS_WKT  0ull
#define WS_WVT  (WS_WKT + 2ull*CC*CC)
#define WS_WPT  (WS_WVT + 2ull*CC*CC)
#define WS_QBF  (WS_WPT + 2ull*CC*CC)                    // [BHC][16][DD] bf16
#define WS_XCLS (WS_QBF + 2ull*BHC*16*DD)                // [NB*TT][CC] f32
#define WS_ATTB (WS_XCLS + 4ull*NB*TT*CC)                // [BHC][16][NN] bf16
#define WS_KBF  (WS_ATTB + 2ull*BHC*16*NN)               // [BHC][NN][DD] bf16
#define WS_VTB  (WS_KBF + 2ull*(size_t)BHC*NN*DD)        // [BHC][DD][NN] bf16

// ---- d_out layout (floats) ----
#define OUT_ATTN ((size_t)NB*TT*CC)
#define OUT_V    (OUT_ATTN + (size_t)NB*HH*TT*NN)

#define WMMA_BF16(a,b,c) __builtin_amdgcn_wmma_f32_16x16x32_bf16(false,(a),false,(b),(short)0,(c),false,false)

static __device__ inline v16bf ld_bfrag(const __bf16* p) {
  v8bf lo = *(const v8bf*)p;
  v8bf hi = *(const v8bf*)(p + 16);
  return __builtin_shufflevector(lo, hi, 0,1,2,3,4,5,6,7,8,9,10,11,12,13,14,15);
}

static __device__ inline v16bf ld_afrag_f32(const float* p) {
  vf4 f0 = *(const vf4*)(p);
  vf4 f1 = *(const vf4*)(p + 4);
  vf4 f2 = *(const vf4*)(p + 16);
  vf4 f3 = *(const vf4*)(p + 20);
  v16bf a;
#pragma unroll
  for (int i = 0; i < 4; ++i) {
    a[i]      = (__bf16)f0[i];
    a[4 + i]  = (__bf16)f1[i];
    a[8 + i]  = (__bf16)f2[i];
    a[12 + i] = (__bf16)f3[i];
  }
  return a;
}

static __device__ inline v8f vzero8() {
  v8f z;
#pragma unroll
  for (int i = 0; i < 8; ++i) z[i] = 0.0f;
  return z;
}

// ---------- kernel 1: transpose Wk/Wv/Wp -> bf16 W^T ----------
__global__ void prep_w(const float* Wk, const float* Wv, const float* Wp, char* ws) {
  int i = blockIdx.x * 256 + threadIdx.x;
  const int per = CC * CC;
  if (i >= 3 * per) return;
  int which = i / per, rem = i - which * per;
  int j = rem / CC, k = rem - j * CC;          // dst[j][k] = src[k][j]
  const float* src = (which == 0) ? Wk : ((which == 1) ? Wv : Wp);
  __bf16* dst = (__bf16*)(ws + (which == 0 ? WS_WKT : (which == 1 ? WS_WVT : WS_WPT)));
  dst[(size_t)j * CC + k] = (__bf16)src[(size_t)k * CC + j];
}

// ---------- kernel 2: q = (x[:, :T] @ Wq) * scale -> bf16, rows padded to 16 ----------
__global__ void proj_q(const float* x, const float* Wq, char* ws) {
  __shared__ float xr[CC];
  int row = blockIdx.y;                 // 0 .. NB*16-1
  int b = row >> 4, t = row & 15;
  int col = blockIdx.x * 256 + threadIdx.x;   // < CC
  __bf16* qb = (__bf16*)(ws + WS_QBF);
  int h = col >> 6, dd = col & 63;
  size_t qi = (((size_t)(b * HH + h)) * 16 + t) * DD + dd;
  if (t >= TT) { qb[qi] = (__bf16)0.0f; return; }  // t uniform per block
  const float* xp = x + ((size_t)b * NN + t) * CC;
  for (int i = threadIdx.x; i < CC; i += 256) xr[i] = xp[i];
  __syncthreads();
  float acc = 0.0f;
  for (int k = 0; k < CC; ++k) acc = fmaf(xr[k], Wq[(size_t)k * CC + col], acc);
  qb[qi] = (__bf16)(acc * 0.125f);      // d^-0.5, d=64
}

// ---------- kernel 3: K/V projection (the big GEMM) ----------
// Workgroup = 128 M-rows x 96 cols (both k and v).  Waves split M (16 rows each)
// so all 8 waves issue identical B (weight) addresses -> WGP$ broadcast, and the
// weight matrices are pulled from L2 once per 128 rows instead of once per 16.
// grid = (8 N-chunks [fast], 2048 M-tiles) so the 8 wgs sharing an x block are
// launch-adjacent and x stays L2-resident (fetched ~once from HBM).
__global__ void __launch_bounds__(256) proj_kv(const float* x, char* ws, float* out) {
  int tid = threadIdx.x;
  int wave = tid >> 5, lane = tid & 31, l16 = lane & 15, half = lane >> 4;
  int koff = half * 8;
  int m0 = (blockIdx.y * 8 + wave) * 16;     // per-wave 16-row strip
  int b  = m0 >> 12;                          // / NN (128-row wg never crosses b)
  int n0 = m0 & (NN - 1);
  int cbase = blockIdx.x * 96;                // shared by all waves in the wg

  const __bf16* wkt = (const __bf16*)(ws + WS_WKT);
  const __bf16* wvt = (const __bf16*)(ws + WS_WVT);
  __bf16* kb = (__bf16*)(ws + WS_KBF);
  __bf16* vt = (__bf16*)(ws + WS_VTB);
  float* vout = out + OUT_V;

  v8f ak[6], av[6];
#pragma unroll
  for (int i = 0; i < 6; ++i) { ak[i] = vzero8(); av[i] = vzero8(); }

  const float* xrow = x + (size_t)(m0 + l16) * CC;
  // prefetch this lane's x row (gfx1250 global_prefetch_b8)
  __builtin_prefetch(xrow, 0, 3);
  __builtin_prefetch(xrow + 384, 0, 3);

  for (int k0 = 0; k0 < CC; k0 += 32) {
    v16bf a = ld_afrag_f32(xrow + k0 + koff);
#pragma unroll
    for (int tt = 0; tt < 6; ++tt) {
      int wr = cbase + tt * 16 + l16;
      v16bf bk = ld_bfrag(wkt + (size_t)wr * CC + k0 + koff);
      ak[tt] = WMMA_BF16(a, bk, ak[tt]);
      v16bf bv = ld_bfrag(wvt + (size_t)wr * CC + k0 + koff);
      av[tt] = WMMA_BF16(a, bv, av[tt]);
    }
  }
#pragma unroll
  for (int tt = 0; tt < 6; ++tt) {
    int col = cbase + tt * 16 + l16;
    int h = col >> 6, dd = col & 63;
    size_t bh = (size_t)b * HH + h;
    v8bf pk;
#pragma unroll
    for (int r = 0; r < 8; ++r) {
      int n = n0 + half * 8 + r;                       // D row
      size_t idx = (bh * NN + n) * DD + dd;
      kb[idx]   = (__bf16)ak[tt][r];
      vout[idx] = av[tt][r];
      pk[r] = (__bf16)av[tt][r];
    }
    size_t vti = (bh * DD + dd) * NN + n0 + half * 8;  // v^T, contiguous in n
    *(v8bf*)(vt + vti) = pk;
  }
}

// ---------- kernel 4: masked attention + softmax + AV, one wg per (b,h) ----------
__global__ void __launch_bounds__(256) attn_k(char* ws, float* out) {
  int bh = blockIdx.x;
  int tid = threadIdx.x;
  int wave = tid >> 5, lane = tid & 31, l16 = lane & 15, half = lane >> 4;
  int koff = half * 8;

  __shared__ float red[8][16];
  __shared__ float rowmax[16];
  __shared__ float rowinv[16];
  __shared__ float part[8][16 * 64];

  const __bf16* qb = (const __bf16*)(ws + WS_QBF) + (size_t)bh * 16 * DD;
  const __bf16* kb = (const __bf16*)(ws + WS_KBF) + (size_t)bh * NN * DD;
  const __bf16* vt = (const __bf16*)(ws + WS_VTB) + (size_t)bh * DD * NN;
  __bf16* ab = (__bf16*)(ws + WS_ATTB) + (size_t)bh * 16 * NN;
  float* attn = out + OUT_ATTN + (size_t)bh * TT * NN;

  // q fragments, cached for the whole kernel
  v16bf aq0 = ld_bfrag(qb + (size_t)l16 * DD + 0 + koff);
  v16bf aq1 = ld_bfrag(qb + (size_t)l16 * DD + 32 + koff);

  int nstart = wave * 512;

  // -- phase 1: logits + mask + running row max --
  float pmax[8];
#pragma unroll
  for (int r = 0; r < 8; ++r) pmax[r] = -3.0e38f;

  for (int t32 = 0; t32 < 32; ++t32) {
    int n0 = nstart + t32 * 16;
    v8f acc = vzero8();
    v16bf b0 = ld_bfrag(kb + (size_t)(n0 + l16) * DD + 0 + koff);
    acc = WMMA_BF16(aq0, b0, acc);
    v16bf b1 = ld_bfrag(kb + (size_t)(n0 + l16) * DD + 32 + koff);
    acc = WMMA_BF16(aq1, b1, acc);
    int n = n0 + l16;
#pragma unroll
    for (int r = 0; r < 8; ++r) {
      int row = r + half * 8;
      if (row < TT) {
        float v = acc[r];
        if (n < TT && n != row) v = -1.0e30f;    // mask: other task tokens
        attn[(size_t)row * NN + n] = v;
        pmax[r] = fmaxf(pmax[r], v);
      }
    }
  }
#pragma unroll
  for (int r = 0; r < 8; ++r) {
    float v = pmax[r];
    for (int off = 1; off < 16; off <<= 1) v = fmaxf(v, __shfl_xor(v, off, 32));
    pmax[r] = v;
  }
  if (l16 == 0) {
#pragma unroll
    for (int r = 0; r < 8; ++r) red[wave][half * 8 + r] = pmax[r];
  }
  __syncthreads();
  if (tid < 16) {
    float m = red[0][tid];
    for (int w = 1; w < 8; ++w) m = fmaxf(m, red[w][tid]);
    rowmax[tid] = m;
  }
  __syncthreads();

  // -- phase 2: exp + row sum --
  float psum[8];
#pragma unroll
  for (int r = 0; r < 8; ++r) psum[r] = 0.0f;
  for (int t32 = 0; t32 < 32; ++t32) {
    int n = nstart + t32 * 16 + l16;
#pragma unroll
    for (int r = 0; r < 8; ++r) {
      int row = r + half * 8;
      if (row < TT) {
        size_t ai = (size_t)row * NN + n;
        float e = __expf(attn[ai] - rowmax[row]);
        attn[ai] = e;
        psum[r] += e;
      }
    }
  }
#pragma unroll
  for (int r = 0; r < 8; ++r) {
    float v = psum[r];
    for (int off = 1; off < 16; off <<= 1) v += __shfl_xor(v, off, 32);
    psum[r] = v;
  }
  if (l16 == 0) {
#pragma unroll
    for (int r = 0; r < 8; ++r) red[wave][half * 8 + r] = psum[r];
  }
  __syncthreads();
  if (tid < 16) {
    float s = 0.0f;
    for (int w = 0; w < 8; ++w) s += red[w][tid];
    rowinv[tid] = 1.0f / s;                       // rows>=TT unused
  }
  __syncthreads();

  // -- phase 3: normalize, write final attn (f32) + bf16 copy for AV --
  for (int t32 = 0; t32 < 32; ++t32) {
    int n = nstart + t32 * 16 + l16;
#pragma unroll
    for (int r = 0; r < 8; ++r) {
      int row = r + half * 8;
      if (row < TT) {
        size_t ai = (size_t)row * NN + n;
        float a = attn[ai] * rowinv[row];
        attn[ai] = a;
        ab[(size_t)row * NN + n] = (__bf16)a;
      } else {
        ab[(size_t)row * NN + n] = (__bf16)0.0f;  // pad rows 10..15
      }
    }
  }

  // -- phase AV: x_cls[16,64] += attn_bf16[16, chunk] @ v[chunk, 64] --
  v8f c4[4];
#pragma unroll
  for (int i = 0; i < 4; ++i) c4[i] = vzero8();
  for (int ks = 0; ks < 16; ++ks) {
    int n0 = nstart + ks * 32;
    v16bf a = ld_bfrag(ab + (size_t)l16 * NN + n0 + koff);
#pragma unroll
    for (int ct = 0; ct < 4; ++ct) {
      int dd = ct * 16 + l16;
      v16bf bv = ld_bfrag(vt + (size_t)dd * NN + n0 + koff);
      c4[ct] = WMMA_BF16(a, bv, c4[ct]);
    }
  }
#pragma unroll
  for (int ct = 0; ct < 4; ++ct) {
    int dd = ct * 16 + l16;
#pragma unroll
    for (int r = 0; r < 8; ++r) {
      int row = r + half * 8;
      part[wave][row * 64 + dd] = c4[ct][r];
    }
  }
  __syncthreads();
  // deterministic cross-wave reduction + write x_cls intermediate [B,T,H,d]
  int b = bh / HH, h = bh - b * HH;
  float* xc = (float*)(ws + WS_XCLS);
  for (int i = tid; i < TT * 64; i += 256) {
    int row = i >> 6, dd = i & 63;
    float s = 0.0f;
    for (int w = 0; w < 8; ++w) s += part[w][row * 64 + dd];
    xc[(((size_t)b * TT + row) * HH + h) * DD + dd] = s;
  }
}

// ---------- kernel 5: x_cls = xcls @ Wp + bp ----------
__global__ void __launch_bounds__(256) proj_p(const float* bp, char* ws, float* out) {
  int tid = threadIdx.x;
  int wave = tid >> 5, lane = tid & 31, l16 = lane & 15, half = lane >> 4;
  int koff = half * 8;
  int m0 = blockIdx.x * 16;             // 40 blocks, 640 rows
  int cbase = wave * 96;
  const float* xc = (const float*)(ws + WS_XCLS);
  const __bf16* wpt = (const __bf16*)(ws + WS_WPT);

  v8f acc[6];
#pragma unroll
  for (int i = 0; i < 6; ++i) acc[i] = vzero8();

  const float* xrow = xc + (size_t)(m0 + l16) * CC;
  for (int k0 = 0; k0 < CC; k0 += 32) {
    v16bf a = ld_afrag_f32(xrow + k0 + koff);
#pragma unroll
    for (int tt = 0; tt < 6; ++tt) {
      int wr = cbase + tt * 16 + l16;
      v16bf b = ld_bfrag(wpt + (size_t)wr * CC + k0 + koff);
      acc[tt] = WMMA_BF16(a, b, acc[tt]);
    }
  }
#pragma unroll
  for (int tt = 0; tt < 6; ++tt) {
    int col = cbase + tt * 16 + l16;
    float bias = bp[col];
#pragma unroll
    for (int r = 0; r < 8; ++r) {
      int row = m0 + r + half * 8;
      out[(size_t)row * CC + col] = acc[tt][r] + bias;
    }
  }
}

extern "C" void kernel_launch(void* const* d_in, const int* in_sizes, int n_in,
                              void* d_out, int out_size, void* d_ws, size_t ws_size,
                              hipStream_t stream) {
  (void)in_sizes; (void)n_in; (void)out_size; (void)ws_size;
  const float* x  = (const float*)d_in[0];
  const float* Wq = (const float*)d_in[1];
  const float* Wk = (const float*)d_in[2];
  const float* Wv = (const float*)d_in[3];
  const float* Wp = (const float*)d_in[4];
  const float* bp = (const float*)d_in[5];
  char* ws = (char*)d_ws;
  float* out = (float*)d_out;

  { int n = 3 * CC * CC;
    prep_w<<<(n + 255) / 256, 256, 0, stream>>>(Wk, Wv, Wp, ws); }
  { dim3 g(CC / 256, NB * 16);
    proj_q<<<g, 256, 0, stream>>>(x, Wq, ws); }
  { dim3 g(8, (NB * NN) / 128);          // (N-chunk fast, M-tile slow)
    proj_kv<<<g, 256, 0, stream>>>(x, ws, out); }
  { attn_k<<<NB * HH, 256, 0, stream>>>(ws, out); }
  { proj_p<<<(NB * TT) / 16, 256, 0, stream>>>(bp, ws, out); }
}
